// Detection_model_79628693668168
// MI455X (gfx1250) — matrix operand order repaired and checked by reading, not compiled, not run on hardware
//
#include <hip/hip_runtime.h>
#include <hip/hip_bf16.h>

// MI455X / gfx1250, wave32. bf16 WMMA (16x16x32, f32 accum) for all three GEMMs
// (x-projection, GRU recurrence, head MLP). GRU recurrence is per-workgroup:
// 16 batch rows per WG, hidden state in LDS (bf16 A-operand) + registers,
// __syncthreads-only per-step sync. x_proj GEMM is 2x2 register-blocked for
// 2x arithmetic intensity out of L2.

typedef __bf16 bf16_t;
typedef bf16_t v16bf __attribute__((ext_vector_type(16)));
typedef bf16_t v8bf  __attribute__((ext_vector_type(8)));
typedef float  v8f   __attribute__((ext_vector_type(8)));

#define B_   128
#define N_   200
#define E_   50
#define D_   768
#define G3_  2304   // 3*D
#define BN_  25600  // B*N
#define NTN_ 144    // G3/16 column tiles
#define F2_  1536   // 2*D (feats width / head1 K)

#define WMMA_BF16(a, b, c) \
  __builtin_amdgcn_wmma_f32_16x16x32_bf16(false, (a), false, (b), (short)0, (c), false, false)

__device__ __forceinline__ float sigmoid_f(float x) {
  return 1.0f / (1.0f + __expf(-x));
}

// ---------------------------------------------------------------------------
// prep: f32->bf16 conversion of x, W_ih, W_hh; W1 transpose->bf16;
//       sum_e encode_event; sum_n w
// ---------------------------------------------------------------------------
__global__ void prep_kernel(const float* __restrict__ enc,
                            const float* __restrict__ x,
                            const float* __restrict__ wsen,
                            const float* __restrict__ Wih,
                            const float* __restrict__ Whh,
                            const float* __restrict__ W1,
                            bf16_t* __restrict__ wihb,
                            bf16_t* __restrict__ whhb,
                            bf16_t* __restrict__ xbf,
                            bf16_t* __restrict__ w1tb,
                            float* __restrict__ sume,
                            float* __restrict__ wsum) {
  int tid = blockIdx.x * blockDim.x + threadIdx.x;
  int nt  = gridDim.x * blockDim.x;
  for (int i = tid; i < G3_ * D_; i += nt) {
    wihb[i] = (bf16_t)Wih[i];
    whhb[i] = (bf16_t)Whh[i];
  }
  for (int i = tid; i < BN_ * D_; i += nt) xbf[i] = (bf16_t)x[i];
  // W1 [1536,768] row-major -> w1t [768,1536] bf16 (K-contiguous B operand)
  for (int i = tid; i < F2_ * D_; i += nt) {
    int r = i / D_, c = i - r * D_;
    w1tb[(size_t)c * F2_ + r] = (bf16_t)W1[i];
  }
  for (int i = tid; i < B_ * D_; i += nt) {
    int b = i / D_, d = i - b * D_;
    const float* p = enc + (size_t)b * E_ * D_ + d;
    float s = 0.f;
    for (int e = 0; e < E_; ++e) s += p[(size_t)e * D_];
    sume[i] = s;
  }
  for (int i = tid; i < B_; i += nt) {
    const float* p = wsen + (size_t)i * N_;
    float s = 0.f;
    for (int n = 0; n < N_; ++n) s += p[n];
    wsum[i] = s;
  }
}

// ---------------------------------------------------------------------------
// x_proj = x @ W_ih^T + b_ih : [25600,768] x [768,2304], f32 out.
// 2x2 register blocking: each wave owns a 32x32 output block (4 WMMA tiles),
// halving L2 traffic per FLOP vs one-tile-per-wave.
// ---------------------------------------------------------------------------
__global__ void __launch_bounds__(256)
xproj_gemm(const bf16_t* __restrict__ xbf,
           const bf16_t* __restrict__ wih,
           const float* __restrict__ bih,
           float* __restrict__ xproj) {
  const int lane = threadIdx.x & 31;
  const int blk  = blockIdx.x * 8 + (threadIdx.x >> 5);  // 57600 32x32 blocks
  const int NB   = NTN_ / 2;                             // 72
  const int m0   = (blk / NB) * 32;
  const int n0   = (blk % NB) * 32;
  const int half = lane >> 4;
  const int l15  = lane & 15;

  // A operand (16-bit 16x32): lane row = l15, lane halves split K 0-15/16-31
  const bf16_t* ap0 = xbf + (size_t)(m0 + l15) * D_ + half * 16;
  const bf16_t* ap1 = ap0 + (size_t)16 * D_;
  // B operand (16-bit 32x16): lane col = l15, lane halves split K 0-15/16-31
  const bf16_t* bp0 = wih + (size_t)(n0 + l15) * D_ + half * 16;
  const bf16_t* bp1 = bp0 + (size_t)16 * D_;

  v8f c00 = {}, c01 = {}, c10 = {}, c11 = {};
#pragma unroll 2
  for (int k = 0; k < D_; k += 32) {
    v8bf a0lo = *(const v8bf*)(ap0 + k);
    v8bf a0hi = *(const v8bf*)(ap0 + k + 8);
    v8bf a1lo = *(const v8bf*)(ap1 + k);
    v8bf a1hi = *(const v8bf*)(ap1 + k + 8);
    v16bf a0, a1;
#pragma unroll
    for (int q = 0; q < 8; ++q) {
      a0[q] = a0lo[q]; a0[q + 8] = a0hi[q];
      a1[q] = a1lo[q]; a1[q + 8] = a1hi[q];
    }
    v16bf b0 = *(const v16bf*)(bp0 + k);
    v16bf b1 = *(const v16bf*)(bp1 + k);
    c00 = WMMA_BF16(a0, b0, c00);
    c01 = WMMA_BF16(a0, b1, c01);
    c10 = WMMA_BF16(a1, b0, c10);
    c11 = WMMA_BF16(a1, b1, c11);
  }
  // C/D layout: VGPR r -> (M = r + 8*half, N = l15)
  const float bias0 = bih[n0 + l15];
  const float bias1 = bih[n0 + 16 + l15];
#pragma unroll
  for (int r = 0; r < 8; ++r) {
    int row0 = m0 + half * 8 + r;
    int row1 = row0 + 16;
    xproj[(size_t)row0 * G3_ + n0 + l15]      = c00[r] + bias0;
    xproj[(size_t)row0 * G3_ + n0 + 16 + l15] = c01[r] + bias1;
    xproj[(size_t)row1 * G3_ + n0 + l15]      = c10[r] + bias0;
    xproj[(size_t)row1 * G3_ + n0 + 16 + l15] = c11[r] + bias1;
  }
}

// ---------------------------------------------------------------------------
// GRU scan. blockIdx.x in [0,8): batch rows m0..m0+15. 1024 threads = 32 waves.
// Wave w owns column tiles {w, w+32} (48 tiles of 16 over D=768).
// Per step, per owned column tile: 3 gate WMMA tiles (r,z,n) sharing one
// LDS A-fragment of h(bf16); gate math + h/acc update in registers.
// Epilogue emits feats = [E*acc - wsum*sume , acc] as bf16 [128][1536].
// ---------------------------------------------------------------------------
__global__ void __launch_bounds__(1024, 1)
gru_scan(const bf16_t* __restrict__ whh,
         const float* __restrict__ bhh,
         const float* __restrict__ xproj,
         const float* __restrict__ wsen,
         const float* __restrict__ sume,
         const float* __restrict__ wsum,
         bf16_t* __restrict__ featsb) {
  __shared__ __align__(16) bf16_t hbuf[16 * D_];  // 24 KB bf16 hidden state

  const int wave = threadIdx.x >> 5;
  const int lane = threadIdx.x & 31;
  const int half = lane >> 4;
  const int l15  = lane & 15;
  const int m0   = blockIdx.x * 16;

  for (int i = threadIdx.x; i < 16 * D_; i += 1024) hbuf[i] = (bf16_t)0.f;
  __syncthreads();

  v8f hreg[2] = {{}, {}};   // hidden state, one 16x16 tile per round
  v8f areg[2] = {{}, {}};   // sum_t w[b,t]*h_t

  int   dcol[2];
  float bhr[2], bhz[2], bhn[2];
#pragma unroll
  for (int r = 0; r < 2; ++r) {
    int jt = wave + r * 32;
    int d  = jt * 16 + l15;
    dcol[r] = d;
    if (jt < 48) {
      bhr[r] = bhh[d];
      bhz[r] = bhh[D_ + d];
      bhn[r] = bhh[2 * D_ + d];
    } else {
      bhr[r] = bhz[r] = bhn[r] = 0.f;
    }
  }

  const bf16_t* aP = hbuf + l15 * D_ + half * 16;  // A-fragment base (LDS)

  for (int t = 0; t < N_; ++t) {
#pragma unroll
    for (int r = 0; r < 2; ++r) {
      int jt = wave + r * 32;
      if (jt >= 48) continue;
      int d = dcol[r];
      const bf16_t* bp0 = whh + (size_t)(d)          * D_ + half * 16;
      const bf16_t* bp1 = whh + (size_t)(D_ + d)     * D_ + half * 16;
      const bf16_t* bp2 = whh + (size_t)(2 * D_ + d) * D_ + half * 16;
      v8f c0 = {}, c1 = {}, c2 = {};
#pragma unroll 2
      for (int k = 0; k < D_; k += 32) {
        v8bf alo = *(const v8bf*)(aP + k);
        v8bf ahi = *(const v8bf*)(aP + k + 8);
        v16bf a;
#pragma unroll
        for (int q = 0; q < 8; ++q) { a[q] = alo[q]; a[q + 8] = ahi[q]; }
        v16bf b0 = *(const v16bf*)(bp0 + k);
        v16bf b1 = *(const v16bf*)(bp1 + k);
        v16bf b2 = *(const v16bf*)(bp2 + k);
        c0 = WMMA_BF16(a, b0, c0);
        c1 = WMMA_BF16(a, b1, c1);
        c2 = WMMA_BF16(a, b2, c2);
      }
      // gate math: element i -> (batch b = m0 + i + 8*half, feature d)
#pragma unroll
      for (int i = 0; i < 8; ++i) {
        int b = m0 + i + half * 8;
        size_t xrow = ((size_t)b * N_ + t) * (size_t)G3_;
        float xr = xproj[xrow + d];
        float xz = xproj[xrow + D_ + d];
        float xn = xproj[xrow + 2 * D_ + d];
        float rg = sigmoid_f(xr + c0[i] + bhr[r]);
        float zg = sigmoid_f(xz + c1[i] + bhz[r]);
        float ng = tanhf(xn + rg * (c2[i] + bhn[r]));
        float hnew = (1.f - zg) * ng + zg * hreg[r][i];
        hreg[r][i] = hnew;
        areg[r][i] += wsen[(size_t)b * N_ + t] * hnew;
      }
    }
    __syncthreads();  // all waves done reading old h from LDS
#pragma unroll
    for (int r = 0; r < 2; ++r) {
      int jt = wave + r * 32;
      if (jt >= 48) continue;
      int d = dcol[r];
#pragma unroll
      for (int i = 0; i < 8; ++i) {
        hbuf[(i + half * 8) * D_ + d] = (bf16_t)hreg[r][i];
      }
    }
    __syncthreads();  // new h visible before next step's GEMM
  }

  // epilogue: feats[b, d] = E*acc - wsum*sume (D_vec), feats[b, D+d] = acc (E_vec)
#pragma unroll
  for (int r = 0; r < 2; ++r) {
    int jt = wave + r * 32;
    if (jt >= 48) continue;
    int d = dcol[r];
#pragma unroll
    for (int i = 0; i < 8; ++i) {
      int b = m0 + i + half * 8;
      float a  = areg[r][i];
      float Dv = (float)E_ * a - wsum[b] * sume[(size_t)b * D_ + d];
      featsb[(size_t)b * F2_ + d]      = (bf16_t)Dv;
      featsb[(size_t)b * F2_ + D_ + d] = (bf16_t)a;
    }
  }
}

// ---------------------------------------------------------------------------
// head1: h1 = relu(feats @ W1 + b1) as bf16 WMMA GEMM [128,1536]x[1536,768].
// B operand is the transposed W1 (w1t [768][1536], K-contiguous).
// ---------------------------------------------------------------------------
__global__ void __launch_bounds__(256)
head1_gemm(const bf16_t* __restrict__ featsb,
           const bf16_t* __restrict__ w1t,
           const float* __restrict__ b1,
           float* __restrict__ h1) {
  const int lane = threadIdx.x & 31;
  const int tile = blockIdx.x * 8 + (threadIdx.x >> 5);  // 8*48 = 384 tiles
  const int m0 = (tile / 48) * 16;
  const int n0 = (tile % 48) * 16;
  const int half = lane >> 4;
  const int l15  = lane & 15;

  const bf16_t* ap = featsb + (size_t)(m0 + l15) * F2_ + half * 16;
  const bf16_t* bp = w1t    + (size_t)(n0 + l15) * F2_ + half * 16;

  v8f c = {};
#pragma unroll 4
  for (int k = 0; k < F2_; k += 32) {
    v8bf alo = *(const v8bf*)(ap + k);
    v8bf ahi = *(const v8bf*)(ap + k + 8);
    v16bf a;
#pragma unroll
    for (int q = 0; q < 8; ++q) { a[q] = alo[q]; a[q + 8] = ahi[q]; }
    v16bf b = *(const v16bf*)(bp + k);
    c = WMMA_BF16(a, b, c);
  }
  const float bias = b1[n0 + l15];
#pragma unroll
  for (int r = 0; r < 8; ++r) {
    int row = m0 + half * 8 + r;
    h1[(size_t)row * D_ + n0 + l15] = fmaxf(c[r] + bias, 0.f);
  }
}

// ---------------------------------------------------------------------------
// head2: out = h1 @ W2 + b2  -> [128, 2] (tiny; scalar f32)
// ---------------------------------------------------------------------------
__global__ void __launch_bounds__(256)
head2_kernel(const float* __restrict__ h1,
             const float* __restrict__ W2,
             const float* __restrict__ b2,
             float* __restrict__ out) {
  int i = blockIdx.x * blockDim.x + threadIdx.x;
  if (i >= B_ * 2) return;
  int b = i >> 1, l = i & 1;
  const float* hp = h1 + (size_t)b * D_;
  float s = b2[l];
  for (int j = 0; j < D_; ++j) s += hp[j] * W2[j * 2 + l];
  out[i] = s;
}

// ---------------------------------------------------------------------------
extern "C" void kernel_launch(void* const* d_in, const int* in_sizes, int n_in,
                              void* d_out, int out_size, void* d_ws, size_t ws_size,
                              hipStream_t stream) {
  (void)in_sizes; (void)n_in; (void)out_size; (void)ws_size;
  const float* enc  = (const float*)d_in[0];
  const float* x    = (const float*)d_in[1];
  const float* wsen = (const float*)d_in[2];
  const float* Wih  = (const float*)d_in[3];
  const float* Whh  = (const float*)d_in[4];
  const float* bih  = (const float*)d_in[5];
  const float* bhh  = (const float*)d_in[6];
  const float* W1   = (const float*)d_in[7];
  const float* b1   = (const float*)d_in[8];
  const float* W2   = (const float*)d_in[9];
  const float* b2   = (const float*)d_in[10];
  float* out = (float*)d_out;

  char* ws = (char*)d_ws;
  auto al = [](size_t v) { return (v + 255) & ~(size_t)255; };
  size_t off = 0;
  bf16_t* wihb   = (bf16_t*)(ws + off); off = al(off + (size_t)G3_ * D_ * 2);
  bf16_t* whhb   = (bf16_t*)(ws + off); off = al(off + (size_t)G3_ * D_ * 2);
  bf16_t* xbf    = (bf16_t*)(ws + off); off = al(off + (size_t)BN_ * D_ * 2);
  bf16_t* w1tb   = (bf16_t*)(ws + off); off = al(off + (size_t)D_ * F2_ * 2);
  bf16_t* featsb = (bf16_t*)(ws + off); off = al(off + (size_t)B_ * F2_ * 2);
  float*  sume   = (float*)(ws + off);  off = al(off + (size_t)B_ * D_ * 4);
  float*  wsum   = (float*)(ws + off);  off = al(off + (size_t)B_ * 4);
  float*  h1     = (float*)(ws + off);  off = al(off + (size_t)B_ * D_ * 4);
  float*  xproj  = (float*)(ws + off);  off = al(off + (size_t)BN_ * G3_ * 4);

  prep_kernel<<<2048, 256, 0, stream>>>(enc, x, wsen, Wih, Whh, W1,
                                        wihb, whhb, xbf, w1tb, sume, wsum);
  // (25600/32)*(2304/32) = 57600 blocks of 32x32, 8 waves/block -> 7200 blocks
  xproj_gemm<<<7200, 256, 0, stream>>>(xbf, wihb, bih, xproj);
  gru_scan<<<8, 1024, 0, stream>>>(whhb, bhh, xproj, wsen, sume, wsum, featsb);
  // 384 head tiles, 8 waves/block -> 48 blocks
  head1_gemm<<<48, 256, 0, stream>>>(featsb, w1tb, b1, h1);
  head2_kernel<<<1, 256, 0, stream>>>(h1, W2, b2, out);
}
